// PPNet_56135222558858
// MI455X (gfx1250) — matrix operand order repaired and checked by reading, not compiled, not run on hardware
//
#include <hip/hip_runtime.h>
#include <hip/hip_bf16.h>

// ---------------------------------------------------------------- types
typedef __attribute__((ext_vector_type(16))) unsigned short v16u;
typedef __attribute__((ext_vector_type(16))) __bf16         v16bf;
typedef __attribute__((ext_vector_type(8)))  float          v8f;

union U4 { uint4 v; unsigned short s[8]; };

__device__ __forceinline__ unsigned short f32_to_bf16(float f) {
  unsigned int u = __float_as_uint(f);
  unsigned int r = (u + 0x7FFFu + ((u >> 16) & 1u)) >> 16;   // RNE
  return (unsigned short)r;
}
__device__ __forceinline__ float bf16_to_f32(unsigned short h) {
  return __uint_as_float(((unsigned int)h) << 16);
}

#define FLAG_RELU 1
#define FLAG_OBF16 2

// ---------------------------------------------------------------- generic conversion
__global__ __launch_bounds__(256) void k_f32_to_bf16(const float* __restrict__ s,
                                                     unsigned short* __restrict__ d, long n) {
  long i = (long)blockIdx.x * 256 + threadIdx.x;
  if (i < n) d[i] = f32_to_bf16(s[i]);
}

// conversion with zero tail padding (dst has ntotal elems, src nsrc elems, same layout)
__global__ __launch_bounds__(256) void k_f32_to_bf16_z(const float* __restrict__ s,
                                                       unsigned short* __restrict__ d,
                                                       long nsrc, long ntotal) {
  long i = (long)blockIdx.x * 256 + threadIdx.x;
  if (i < ntotal) d[i] = (i < nsrc) ? f32_to_bf16(s[i]) : (unsigned short)0;
}

// ---------------------------------------------------------------- positional encoding (64 x 256)
__global__ __launch_bounds__(256) void k_pe(float* __restrict__ pe) {
  int idx = blockIdx.x * 256 + threadIdx.x;          // 0..16383
  int t = idx >> 8, d = idx & 255;
  int i = d >> 1;
  float div = __expf((float)(2 * i) * (-logf(10000.0f) / 256.0f));
  float ang = (float)t * div;
  pe[idx] = (d & 1) ? cosf(ang) : sinf(ang);
}

// ---------------------------------------------------------------- gate = F * softmax(mkt @ gw + gb)
__global__ __launch_bounds__(256) void k_gate(const float* __restrict__ mf,
                                              const float* __restrict__ gw,
                                              const float* __restrict__ gb,
                                              float* __restrict__ gate,
                                              int Ff, int Mf, int T) {
  __shared__ float red[256];
  __shared__ float mrow[64];
  int n = blockIdx.x, tid = threadIdx.x;
  if (tid < Mf) mrow[tid] = mf[((long)n * T + (T - 1)) * Mf + tid];
  __syncthreads();
  float logit = -1e30f;
  if (tid < Ff) {
    float a = gb[tid];
    for (int m = 0; m < Mf; ++m) a += mrow[m] * gw[m * Ff + tid];
    logit = a;
  }
  red[tid] = logit; __syncthreads();
  for (int st = 128; st; st >>= 1) { if (tid < st) red[tid] = fmaxf(red[tid], red[tid + st]); __syncthreads(); }
  float mx = red[0]; __syncthreads();
  float e = (tid < Ff) ? __expf(logit - mx) : 0.0f;
  red[tid] = e; __syncthreads();
  for (int st = 128; st; st >>= 1) { if (tid < st) red[tid] += red[tid + st]; __syncthreads(); }
  float inv = (float)Ff / red[0];
  if (tid < Ff) gate[(long)n * Ff + tid] = e * inv;
}

// ---------------------------------------------------------------- gated feature staging -> bf16 (NT x 160, K padded)
__global__ __launch_bounds__(256) void k_stage_gated(const float* __restrict__ sf,
                                                     const float* __restrict__ gate,
                                                     unsigned short* __restrict__ out,
                                                     long total, int T, int F, int Fp) {
  long i = (long)blockIdx.x * 256 + threadIdx.x;     // over NT * Fp
  if (i >= total) return;
  long row = i / Fp;
  int f = (int)(i % Fp);
  long n = row / T;
  unsigned short o = 0;
  if (f < F) o = f32_to_bf16(sf[row * F + f] * gate[n * F + f]);
  out[i] = o;
}

// ---------------------------------------------------------------- row LayerNorm -> bf16 (wave per row)
__global__ __launch_bounds__(256) void k_ln_rows(const float* __restrict__ in, int in_stride,
                                                 const float* __restrict__ sc,
                                                 const float* __restrict__ bi,
                                                 unsigned short* __restrict__ out, long out_stride,
                                                 long R, int W) {
  long gw = ((long)blockIdx.x * blockDim.x + threadIdx.x) >> 5;
  int lane = threadIdx.x & 31;
  if (gw >= R) return;
  const float* row = in + gw * in_stride;
  int per = W >> 5;                      // W is 32..256
  float x[8];
  float sum = 0.0f;
  for (int j = 0; j < per; ++j) { x[j] = row[lane + j * 32]; sum += x[j]; }
  for (int off = 16; off; off >>= 1) sum += __shfl_xor(sum, off, 32);
  float mean = sum / (float)W;
  float vs = 0.0f;
  for (int j = 0; j < per; ++j) { float d = x[j] - mean; vs += d * d; }
  for (int off = 16; off; off >>= 1) vs += __shfl_xor(vs, off, 32);
  float rstd = rsqrtf(vs / (float)W + 1e-5f);
  for (int j = 0; j < per; ++j) {
    int c = lane + j * 32;
    out[gw * out_stride + c] = f32_to_bf16((x[j] - mean) * rstd * sc[c] + bi[c]);
  }
}

// ---------------------------------------------------------------- WMMA GEMM: C = A(bf16 MxK) @ B(bf16 KxN)
// REQUIRES: Mdim%128==0, Ndim%128==0, Kdim%32==0, lda%8==0, ldb%8==0 (no guards!)
// Double-buffered LDS pipeline: one barrier per k-step, global loads overlap WMMA.
__global__ __launch_bounds__(256) void k_wmma_gemm(const unsigned short* __restrict__ A, int lda,
                                                   const unsigned short* __restrict__ B, int ldb,
                                                   const float* __restrict__ bias,
                                                   const float* __restrict__ resid,
                                                   void* __restrict__ Cout,
                                                   int Mdim, int Ndim, int Kdim,
                                                   int flags, const float* __restrict__ pe) {
  __shared__ __align__(16) unsigned short As[2][128][32];   // 16 KB
  __shared__ __align__(16) unsigned short Bt[2][128][32];   // 16 KB (B transposed: Bt[n][k])
  const int tid = threadIdx.x;
  const int wave = tid >> 5, lane = tid & 31;
  const int ln = lane & 15, hi = lane >> 4;
  const int m0 = blockIdx.x * 128;
  const int n0 = blockIdx.y * 128;
  const int wm = wave >> 1;   // 0..3 -> 32-row band
  const int wn = wave & 1;    // 0..1 -> 64-col band

  // load addressing (all vectorized, no guards)
  const int arow = tid >> 1, acb = (tid & 1) * 16;       // A: 128 rows x 32, 16 elems/thread
  const int bkr = tid >> 3, bnb = (tid & 7) * 16;        // B: 32 rows x 128, 16 elems/thread
  const unsigned short* Aptr = A + (long)(m0 + arow) * lda + acb;
  const unsigned short* Bptr = B + (long)bkr * ldb + n0 + bnb;

  v8f c[2][4] = {};

  const int nsteps = Kdim >> 5;

  // ---- prologue: stage tile 0 into buffer 0
  {
    const uint4* pa = (const uint4*)(Aptr);
    uint4 a0 = pa[0], a1 = pa[1];
    *(uint4*)(&As[0][arow][acb])     = a0;
    *(uint4*)(&As[0][arow][acb + 8]) = a1;
    const uint4* pb = (const uint4*)(Bptr);
    U4 b0, b1; b0.v = pb[0]; b1.v = pb[1];
    #pragma unroll
    for (int j = 0; j < 8; ++j) {
      Bt[0][bnb + j][bkr]     = b0.s[j];
      Bt[0][bnb + 8 + j][bkr] = b1.s[j];
    }
  }
  __syncthreads();

  for (int s = 0; s < nsteps; ++s) {
    const int cur = s & 1, nxt = cur ^ 1;

    // ---- issue next tile's global loads early (overlap with WMMA below)
    uint4 na0, na1; U4 nb0, nb1;
    const bool havenext = (s + 1) < nsteps;
    if (havenext) {
      int k1 = (s + 1) << 5;
      const uint4* pa = (const uint4*)(Aptr + k1);
      na0 = pa[0]; na1 = pa[1];
      const uint4* pb = (const uint4*)(Bptr + (long)k1 * ldb);
      nb0.v = pb[0]; nb1.v = pb[1];
    }

    // ---- fragments from current buffer + 8 WMMAs
    v16bf a[2], b[4];
    #pragma unroll
    for (int fr = 0; fr < 2; ++fr) {      // A frag 16x32 per ISA layout
      int mb = wm * 32 + fr * 16 + ln;
      v16u u;
      #pragma unroll
      for (int v = 0; v < 8; ++v) {
        int kk = ((v < 4) ? (2 * v) : (16 + 2 * (v - 4))) + 8 * hi;
        u[2 * v]     = As[cur][mb][kk];
        u[2 * v + 1] = As[cur][mb][kk + 1];
      }
      a[fr] = __builtin_bit_cast(v16bf, u);
    }
    #pragma unroll
    for (int fc = 0; fc < 4; ++fc) {      // B frag 32x16 per ISA layout
      int nb = wn * 64 + fc * 16 + ln;
      v16u u;
      #pragma unroll
      for (int v = 0; v < 8; ++v) {
        int kk = ((v < 4) ? (2 * v) : (16 + 2 * (v - 4))) + 8 * hi;
        u[2 * v]     = Bt[cur][nb][kk];
        u[2 * v + 1] = Bt[cur][nb][kk + 1];
      }
      b[fc] = __builtin_bit_cast(v16bf, u);
    }
    #pragma unroll
    for (int fr = 0; fr < 2; ++fr)
      #pragma unroll
      for (int fc = 0; fc < 4; ++fc)
        c[fr][fc] = __builtin_amdgcn_wmma_f32_16x16x32_bf16(
            false, a[fr], false, b[fc], (short)0, c[fr][fc], false, false);

    // ---- commit next tile into the other buffer, single barrier
    if (havenext) {
      *(uint4*)(&As[nxt][arow][acb])     = na0;
      *(uint4*)(&As[nxt][arow][acb + 8]) = na1;
      #pragma unroll
      for (int j = 0; j < 8; ++j) {
        Bt[nxt][bnb + j][bkr]     = nb0.s[j];
        Bt[nxt][bnb + 8 + j][bkr] = nb1.s[j];
      }
    }
    __syncthreads();
  }

  // epilogue: D layout -> row = v + 8*hi, col = lane%16
  #pragma unroll
  for (int fr = 0; fr < 2; ++fr)
    #pragma unroll
    for (int fc = 0; fc < 4; ++fc)
      #pragma unroll
      for (int v = 0; v < 8; ++v) {
        int row = m0 + wm * 32 + fr * 16 + v + 8 * hi;
        int col = n0 + wn * 64 + fc * 16 + ln;
        float val = c[fr][fc][v];
        if (bias)  val += bias[col];
        if (resid) val += resid[(long)row * Ndim + col];
        if (pe)    val += pe[(long)(row & 63) * Ndim + col];
        if (flags & FLAG_RELU) val = fmaxf(val, 0.0f);
        if (flags & FLAG_OBF16) ((unsigned short*)Cout)[(long)row * Ndim + col] = f32_to_bf16(val);
        else                    ((float*)Cout)[(long)row * Ndim + col] = val;
      }
}

// ---------------------------------------------------------------- temporal attention per (n, head): 64x64, hd=64
__global__ __launch_bounds__(256) void k_attn_temporal(const unsigned short* __restrict__ Qg,
                                                       const unsigned short* __restrict__ Kg,
                                                       const unsigned short* __restrict__ Vg,
                                                       unsigned short* __restrict__ Og) {
  __shared__ __align__(16) unsigned short Qs[64][64];   // q rows; later reused for probs bf16
  __shared__ __align__(16) unsigned short Ks[64][64];   // k rows (== Bt for scores)
  __shared__ __align__(16) unsigned short Vt[64][64];   // v transposed (Vt[d][s] == Bt for P@V)
  __shared__ float          Ps[64][64];                  // f32 scores / probs
  const int tid = threadIdx.x;
  const int n = blockIdx.x >> 2, h = blockIdx.x & 3;
  const long base = (long)n * 64 * 256 + h * 64;

  { // load 64x64 tiles of q/k/v (b128 loads)
    int row = tid >> 2, cb = (tid & 3) << 4;
    long g = base + (long)row * 256 + cb;
    const uint4* q4 = (const uint4*)(Qg + g);
    const uint4* k4 = (const uint4*)(Kg + g);
    const uint4* v4 = (const uint4*)(Vg + g);
    *(uint4*)(&Qs[row][cb])     = q4[0];
    *(uint4*)(&Qs[row][cb + 8]) = q4[1];
    *(uint4*)(&Ks[row][cb])     = k4[0];
    *(uint4*)(&Ks[row][cb + 8]) = k4[1];
    U4 v0, v1; v0.v = v4[0]; v1.v = v4[1];
    #pragma unroll
    for (int j = 0; j < 8; ++j) {
      Vt[cb + j][row]     = v0.s[j];
      Vt[cb + 8 + j][row] = v1.s[j];
    }
  }
  __syncthreads();

  const int wave = tid >> 5, lane = tid & 31;
  const int ln = lane & 15, hi = lane >> 4;

  // scores = Q @ K^T / 8  : 4x4 frag tiles, 2 per wave, K-dim 64 = 2 steps
  #pragma unroll
  for (int i = 0; i < 2; ++i) {
    int tt = wave * 2 + i;
    int tr = tt >> 2, tc = tt & 3;
    v8f acc = {};
    #pragma unroll
    for (int ks = 0; ks < 2; ++ks) {
      v16u ua, ub;
      #pragma unroll
      for (int v = 0; v < 8; ++v) {
        int kk = ks * 32 + ((v < 4) ? (2 * v) : (16 + 2 * (v - 4))) + 8 * hi;
        ua[2 * v] = Qs[tr * 16 + ln][kk]; ua[2 * v + 1] = Qs[tr * 16 + ln][kk + 1];
      }
      #pragma unroll
      for (int v = 0; v < 8; ++v) {
        int kk = ks * 32 + ((v < 4) ? (2 * v) : (16 + 2 * (v - 4))) + 8 * hi;
        ub[2 * v] = Ks[tc * 16 + ln][kk]; ub[2 * v + 1] = Ks[tc * 16 + ln][kk + 1];
      }
      acc = __builtin_amdgcn_wmma_f32_16x16x32_bf16(
          false, __builtin_bit_cast(v16bf, ua), false, __builtin_bit_cast(v16bf, ub),
          (short)0, acc, false, false);
    }
    #pragma unroll
    for (int v = 0; v < 8; ++v)
      Ps[tr * 16 + v + 8 * hi][tc * 16 + ln] = acc[v] * 0.125f;
  }
  __syncthreads();

  // row softmax, probs -> bf16 into Qs
  if (tid < 64) {
    float mx = -1e30f;
    for (int s = 0; s < 64; ++s) mx = fmaxf(mx, Ps[tid][s]);
    float sum = 0.0f;
    for (int s = 0; s < 64; ++s) { float e = __expf(Ps[tid][s] - mx); Ps[tid][s] = e; sum += e; }
    float inv = 1.0f / sum;
    for (int s = 0; s < 64; ++s) Qs[tid][s] = f32_to_bf16(Ps[tid][s] * inv);
  }
  __syncthreads();

  // O = P @ V
  #pragma unroll
  for (int i = 0; i < 2; ++i) {
    int tt = wave * 2 + i;
    int tr = tt >> 2, tc = tt & 3;
    v8f acc = {};
    #pragma unroll
    for (int ks = 0; ks < 2; ++ks) {
      v16u ua, ub;
      #pragma unroll
      for (int v = 0; v < 8; ++v) {
        int kk = ks * 32 + ((v < 4) ? (2 * v) : (16 + 2 * (v - 4))) + 8 * hi;
        ua[2 * v] = Qs[tr * 16 + ln][kk]; ua[2 * v + 1] = Qs[tr * 16 + ln][kk + 1];
      }
      #pragma unroll
      for (int v = 0; v < 8; ++v) {
        int kk = ks * 32 + ((v < 4) ? (2 * v) : (16 + 2 * (v - 4))) + 8 * hi;
        ub[2 * v] = Vt[tc * 16 + ln][kk]; ub[2 * v + 1] = Vt[tc * 16 + ln][kk + 1];
      }
      acc = __builtin_amdgcn_wmma_f32_16x16x32_bf16(
          false, __builtin_bit_cast(v16bf, ua), false, __builtin_bit_cast(v16bf, ub),
          (short)0, acc, false, false);
    }
    #pragma unroll
    for (int v = 0; v < 8; ++v) {
      int t = tr * 16 + v + 8 * hi;
      int d = tc * 16 + ln;
      Og[base + (long)t * 256 + d] = f32_to_bf16(acc[v]);
    }
  }
}

// ---------------------------------------------------------------- temporal aggregation (block per n)
__global__ __launch_bounds__(256) void k_tagg(const float* __restrict__ xt,
                                              const float* __restrict__ h,
                                              float* __restrict__ z, int T, int D) {
  __shared__ float part[256];
  __shared__ float lam[64];
  int n = blockIdx.x, tid = threadIdx.x;
  const float* hb = h + (long)n * T * D;
  const float* hl = hb + (long)(T - 1) * D;
  int t = tid >> 2, p = tid & 3;
  float s = 0.0f;
  for (int j = 0; j < 64; ++j) s += hb[(long)t * D + p * 64 + j] * hl[p * 64 + j];
  part[tid] = s; __syncthreads();
  if (p == 0) lam[t] = part[tid] + part[tid + 1] + part[tid + 2] + part[tid + 3];
  __syncthreads();
  if (tid == 0) {
    float mx = -1e30f;
    for (int i = 0; i < 64; ++i) mx = fmaxf(mx, lam[i]);
    float sum = 0.0f;
    for (int i = 0; i < 64; ++i) { lam[i] = __expf(lam[i] - mx); sum += lam[i]; }
    float inv = 1.0f / sum;
    for (int i = 0; i < 64; ++i) lam[i] *= inv;
  }
  __syncthreads();
  const float* xb = xt + (long)n * T * D;
  float acc = 0.0f;
  for (int tt = 0; tt < T; ++tt) acc += lam[tt] * xb[(long)tt * D + tid];
  z[(long)n * D + tid] = acc;
}

// ---------------------------------------------------------------- industry embedding gather
__global__ __launch_bounds__(256) void k_gather_emb(const int* __restrict__ ids,
                                                    const float* __restrict__ table,
                                                    float* __restrict__ out, long total, int E) {
  long i = (long)blockIdx.x * 256 + threadIdx.x;
  if (i >= total) return;
  out[i] = table[(long)ids[i / E] * E + (int)(i % E)];
}

// ---------------------------------------------------------------- cross-sectional flash attention (wave per (n,h))
__global__ __launch_bounds__(128) void k_xattn(const float* __restrict__ SQ,
                                               const float* __restrict__ SK,
                                               const float* __restrict__ SV,
                                               unsigned short* __restrict__ SO, int Ns) {
  int wave = threadIdx.x >> 5, lane = threadIdx.x & 31;
  int gw = blockIdx.x * 4 + wave;
  int n = gw >> 2, h = gw & 3;
  const float* q = SQ + (long)n * 256 + h * 64;
  float qr[64];
  #pragma unroll
  for (int d = 0; d < 64; ++d) qr[d] = q[d];
  float m = -1e30f, s = 0.0f;
  float acc[64];
  #pragma unroll
  for (int d = 0; d < 64; ++d) acc[d] = 0.0f;

  for (int mb = lane; mb < Ns; mb += 32) {
    const float* kk = SK + (long)mb * 256 + h * 64;
    float sc = 0.0f;
    #pragma unroll
    for (int d = 0; d < 64; ++d) sc += qr[d] * kk[d];
    sc *= 0.125f;
    float mnew = fmaxf(m, sc);
    float corr = __expf(m - mnew);
    float p = __expf(sc - mnew);
    s = s * corr + p;
    const float* vv = SV + (long)mb * 256 + h * 64;
    #pragma unroll
    for (int d = 0; d < 64; ++d) acc[d] = acc[d] * corr + p * vv[d];
    m = mnew;
  }
  // merge lanes
  float M = m;
  for (int off = 16; off; off >>= 1) M = fmaxf(M, __shfl_xor(M, off, 32));
  float f = __expf(m - M);
  float sw = s * f;
  for (int off = 16; off; off >>= 1) sw += __shfl_xor(sw, off, 32);
  float inv = 1.0f / sw;
  for (int d = 0; d < 64; ++d) {
    float a = acc[d] * f;
    for (int off = 16; off; off >>= 1) a += __shfl_xor(a, off, 32);
    if (lane == 0) SO[(long)n * 256 + h * 64 + d] = f32_to_bf16(a * inv);
  }
}

// ---------------------------------------------------------------- decoder (wave per row)
__global__ __launch_bounds__(256) void k_dec(const float* __restrict__ zt,
                                             const float* __restrict__ w,
                                             const float* __restrict__ b,
                                             float* __restrict__ out, int R) {
  long gw = ((long)blockIdx.x * blockDim.x + threadIdx.x) >> 5;
  int lane = threadIdx.x & 31;
  if (gw >= R) return;
  const float* r = zt + gw * 256;
  float s = 0.0f;
  #pragma unroll
  for (int j = 0; j < 8; ++j) s += r[lane + j * 32] * w[lane + j * 32];
  for (int off = 16; off; off >>= 1) s += __shfl_xor(s, off, 32);
  if (lane == 0) out[gw] = s + b[0];
}

// ================================================================ host
extern "C" void kernel_launch(void* const* d_in, const int* in_sizes, int n_in,
                              void* d_out, int out_size, void* d_ws, size_t ws_size,
                              hipStream_t stream) {
  (void)in_sizes; (void)n_in; (void)out_size; (void)ws_size;
  const int N = 4096, T = 64, F = 158, Fp = 160, Mf = 64, E = 32, D = 256;
  const long NT = (long)N * T;

  const int*   industry_ids = (const int*)  d_in[0];
  const float* stock_feat   = (const float*)d_in[1];
  const float* market_feat  = (const float*)d_in[2];
  const float* emb_table    = (const float*)d_in[3];
  const float* gate_w = (const float*)d_in[4];  const float* gate_b = (const float*)d_in[5];
  const float* proj_w = (const float*)d_in[6];  const float* proj_b = (const float*)d_in[7];
  const float* t_qw = (const float*)d_in[8];  const float* t_kw = (const float*)d_in[9];
  const float* t_vw = (const float*)d_in[10]; const float* t_ow = (const float*)d_in[11];
  const float* t_ob = (const float*)d_in[12];
  const float* t_nx_s = (const float*)d_in[13]; const float* t_nx_b = (const float*)d_in[14];
  const float* t_nf_s = (const float*)d_in[15]; const float* t_nf_b = (const float*)d_in[16];
  const float* t_f1w = (const float*)d_in[17]; const float* t_f1b = (const float*)d_in[18];
  const float* t_f2w = (const float*)d_in[19]; const float* t_f2b = (const float*)d_in[20];
  const float* s_qw = (const float*)d_in[21]; const float* s_kw = (const float*)d_in[22];
  const float* s_vw = (const float*)d_in[23]; const float* s_ow = (const float*)d_in[24];
  const float* s_ob = (const float*)d_in[25];
  const float* s_nx_s = (const float*)d_in[26]; const float* s_nx_b = (const float*)d_in[27];
  const float* s_ni_s = (const float*)d_in[28]; const float* s_ni_b = (const float*)d_in[29];
  const float* s_nf_s = (const float*)d_in[30]; const float* s_nf_b = (const float*)d_in[31];
  const float* s_f1w = (const float*)d_in[32]; const float* s_f1b = (const float*)d_in[33];
  const float* s_f2w = (const float*)d_in[34]; const float* s_f2b = (const float*)d_in[35];
  const float* ta_w  = (const float*)d_in[36];
  const float* dec_w = (const float*)d_in[37]; const float* dec_b = (const float*)d_in[38];

  // ---------------- workspace carve-up
  char* ws = (char*)d_ws;
  long off = 0;
  auto alloc = [&](long bytes) -> char* {
    char* p = ws + off; off += (bytes + 255) & ~255L; return p;
  };
  float*          xbuf = (float*)         alloc(NT * D * 4);   // residual stream (f32)
  unsigned short* abuf = (unsigned short*)alloc(NT * D * 2);   // LN outputs / xt_bf16
  unsigned short* qbuf = (unsigned short*)alloc(NT * D * 2);   // q bf16; also gated-A staging; h f32 spans q+k
  unsigned short* kbuf = (unsigned short*)alloc(NT * D * 2);
  unsigned short* vbuf = (unsigned short*)alloc(NT * D * 2);
  unsigned short* obuf = (unsigned short*)alloc(NT * D * 2);   // attn out; later ffn1 out
  float*          hbuf = (float*)qbuf;                          // NT*D f32 overlaps q+k (consumed by then)

  float* pe   = (float*)alloc(64L * 256 * 4);
  float* gate = (float*)alloc((long)N * F * 4);
  float* zbuf = (float*)alloc((long)N * D * 4);
  unsigned short* qkin = (unsigned short*)alloc((long)N * 288 * 2);
  float* sqf = (float*)alloc((long)N * D * 4);
  float* skf = (float*)alloc((long)N * D * 4);
  float* svf = (float*)alloc((long)N * D * 4);
  unsigned short* sob  = (unsigned short*)alloc((long)N * D * 2);
  float* ztbuf = (float*)alloc((long)N * D * 4);
  unsigned short* ynb  = (unsigned short*)alloc((long)N * D * 2);
  unsigned short* sffn = (unsigned short*)alloc((long)N * D * 2);
  float* iemb = (float*)alloc((long)N * E * 4);

  auto cvt = [&](const float* src, long cnt) -> unsigned short* {
    unsigned short* dst = (unsigned short*)alloc(cnt * 2);
    k_f32_to_bf16<<<(unsigned)((cnt + 255) / 256), 256, 0, stream>>>(src, dst, cnt);
    return dst;
  };
  // proj_w: K padded 158 -> 160 with zero rows
  unsigned short* projw = (unsigned short*)alloc((long)Fp * D * 2);
  {
    long nt = (long)Fp * D;
    k_f32_to_bf16_z<<<(unsigned)((nt + 255) / 256), 256, 0, stream>>>(proj_w, projw,
                                                                      (long)F * D, nt);
  }
  unsigned short* tqw   = cvt(t_qw,  (long)D * D);
  unsigned short* tkw   = cvt(t_kw,  (long)D * D);
  unsigned short* tvw   = cvt(t_vw,  (long)D * D);
  unsigned short* tow   = cvt(t_ow,  (long)D * D);
  unsigned short* tf1w  = cvt(t_f1w, (long)D * D);
  unsigned short* tf2w  = cvt(t_f2w, (long)D * D);
  unsigned short* taw   = cvt(ta_w,  (long)D * D);
  unsigned short* sqw   = cvt(s_qw,  (long)(D + E) * D);
  unsigned short* skw   = cvt(s_kw,  (long)(D + E) * D);
  unsigned short* svw   = cvt(s_vw,  (long)D * D);
  unsigned short* sow   = cvt(s_ow,  (long)D * D);
  unsigned short* sf1w  = cvt(s_f1w, (long)D * D);
  unsigned short* sf2w  = cvt(s_f2w, (long)D * D);

  auto gemm = [&](const unsigned short* A, int lda, const unsigned short* B, int ldb,
                  const float* bias, const float* resid, void* C,
                  long Md, int Nd, int Kd, int flags, const float* peptr) {
    dim3 g((unsigned)(Md / 128), (unsigned)(Nd / 128));
    k_wmma_gemm<<<g, 256, 0, stream>>>(A, lda, B, ldb, bias, resid, C,
                                       (int)Md, Nd, Kd, flags, peptr);
  };

  // 1. input gating & projection
  k_pe<<<64, 256, 0, stream>>>(pe);
  k_gate<<<N, 256, 0, stream>>>(market_feat, gate_w, gate_b, gate, F, Mf, T);
  {
    long tot = NT * Fp;
    k_stage_gated<<<(unsigned)((tot + 255) / 256), 256, 0, stream>>>(
        stock_feat, gate, qbuf, tot, T, F, Fp);
  }
  gemm(qbuf, Fp, projw, D, proj_b, nullptr, xbuf, NT, D, Fp, 0, pe);  // x = gated@proj_w + b + pe

  // 2. temporal transformer block
  {
    unsigned lnBlocks = (unsigned)((NT + 7) / 8);
    k_ln_rows<<<lnBlocks, 256, 0, stream>>>(xbuf, D, t_nx_s, t_nx_b, abuf, D, NT, D);
    gemm(abuf, D, tqw, D, nullptr, nullptr, qbuf, NT, D, D, FLAG_OBF16, nullptr);
    gemm(abuf, D, tkw, D, nullptr, nullptr, kbuf, NT, D, D, FLAG_OBF16, nullptr);
    gemm(abuf, D, tvw, D, nullptr, nullptr, vbuf, NT, D, D, FLAG_OBF16, nullptr);
    k_attn_temporal<<<N * 4, 256, 0, stream>>>(qbuf, kbuf, vbuf, obuf);
    gemm(obuf, D, tow, D, t_ob, xbuf, xbuf, NT, D, D, 0, nullptr);    // xt = x + o@t_ow + b
    k_ln_rows<<<lnBlocks, 256, 0, stream>>>(xbuf, D, t_nf_s, t_nf_b, abuf, D, NT, D);
    gemm(abuf, D, tf1w, D, t_f1b, nullptr, obuf, NT, D, D, FLAG_RELU | FLAG_OBF16, nullptr);
    gemm(obuf, D, tf2w, D, t_f2b, xbuf, xbuf, NT, D, D, 0, nullptr);  // xt += ffn
  }

  // 3. temporal aggregation
  {
    long tot = NT * D;
    k_f32_to_bf16<<<(unsigned)((tot + 255) / 256), 256, 0, stream>>>(xbuf, abuf, tot);
    gemm(abuf, D, taw, D, nullptr, nullptr, hbuf, NT, D, D, 0, nullptr);  // h = xt @ ta_w (f32, overlays q/k)
    k_tagg<<<N, 256, 0, stream>>>(xbuf, hbuf, zbuf, T, D);
  }

  // 4. cross-sectional attention
  {
    long totE = (long)N * E;
    k_gather_emb<<<(unsigned)((totE + 255) / 256), 256, 0, stream>>>(
        industry_ids, emb_table, iemb, totE, E);
    unsigned lnN = (unsigned)((N + 7) / 8);
    k_ln_rows<<<lnN, 256, 0, stream>>>(zbuf, D, s_nx_s, s_nx_b, qkin, 288, N, D);        // zs -> cols 0..255
    k_ln_rows<<<lnN, 256, 0, stream>>>(iemb, E, s_ni_s, s_ni_b, qkin + 256, 288, N, E);  // ins -> cols 256..287
    gemm(qkin, 288, sqw, D, nullptr, nullptr, sqf, N, D, D + E, 0, nullptr);
    gemm(qkin, 288, skw, D, nullptr, nullptr, skf, N, D, D + E, 0, nullptr);
    gemm(qkin, 288, svw, D, nullptr, nullptr, svf, N, D, D, 0, nullptr);  // zs only (first 256 cols)
    k_xattn<<<N, 128, 0, stream>>>(sqf, skf, svf, sob, N);                 // N*4 waves, 4 per block
    gemm(sob, D, sow, D, s_ob, zbuf, ztbuf, N, D, D, 0, nullptr);          // zt = z + so@s_ow + b
    k_ln_rows<<<lnN, 256, 0, stream>>>(ztbuf, D, s_nf_s, s_nf_b, ynb, D, N, D);
    gemm(ynb, D, sf1w, D, s_f1b, nullptr, sffn, N, D, D, FLAG_RELU | FLAG_OBF16, nullptr);
    gemm(sffn, D, sf2w, D, s_f2b, ztbuf, ztbuf, N, D, D, 0, nullptr);      // zt += ffn
  }

  // 5. decoder
  k_dec<<<(N * 32 + 255) / 256, 256, 0, stream>>>(ztbuf, dec_w, dec_b, (float*)d_out, N);
}
